// MemoryModule_88759794139458
// MI455X (gfx1250) — matrix operand order repaired and checked by reading, not compiled, not run on hardware
//
#include <hip/hip_runtime.h>
#include <hip/hip_bf16.h>

typedef __attribute__((ext_vector_type(8)))  _Float16 v8h;
typedef __attribute__((ext_vector_type(16))) _Float16 v16h;
typedef __attribute__((ext_vector_type(8)))  float    v8f;

#define B_DIM    64
#define C_DIM    768
#define P_DIM    2304
#define ROWS_T   (B_DIM * P_DIM)          // 147456
#define N_MEM    100
#define NPAD     112
#define M_TILE   128
#define NT1      7                        // 112/16 N-tiles for GEMM1
#define ZK       32                       // K elements staged per step
#define NSTAGE   (C_DIM / ZK)             // 24
#define CCHUNK   128
#define NCHUNK   (C_DIM / CCHUNK)         // 6

#define TEMP_INV (1.0f / 0.07f)
#define LAM_C    (1.0f / 100.0f)
#define EPS_C    1e-8f

// LDS pitches (elements). Chosen so row pitch in dwords = 4 mod 16 -> conflict-free b128.
#define SC_PITCH 116     // scores, f32
#define ZS_PITCH 40      // staged z, f16
#define WH_PITCH 136     // w_hat tile, f16
#define ZH_PITCH 132     // z_hat tile, f32
#define MN_PITCH 768     // global m_norm, f16
#define MT_PITCH 128     // global m_norm^T, f16

#define SCORES_BYTES (M_TILE * SC_PITCH * 4)        // 59392
#define ZS_BYTES     (2 * M_TILE * ZS_PITCH * 2)    // 20480
#define REGION_BYTES (SCORES_BYTES + ZS_BYTES)      // 79872 (>= 67584 z_hat bytes)

// workspace layout (bytes)
#define WS_MEMN_OFF  0
#define WS_MEMT_OFF  (NPAD * MN_PITCH * 2)                   // 172032
#define WS_NORM_OFF  (WS_MEMT_OFF + C_DIM * MT_PITCH * 2)    // 368640 (needs ~369KB of ws)

// A fragment, 16-bit 16x32: lane=M; contiguous K sel16+0..15 per lane
__device__ __forceinline__ v16h frag_a(const _Float16* p) {
  v8h lo = *(const v8h*)(p);
  v8h hi = *(const v8h*)(p + 8);
  return __builtin_shufflevector(lo, hi, 0,1,2,3,4,5,6,7,8,9,10,11,12,13,14,15);
}
// B fragment, 16-bit 32x16: lane=N; contiguous K sel16+0..15 per lane
__device__ __forceinline__ v16h frag_b(const _Float16* p) {
  return *(const v16h*)(p);
}

// ---------------- prep kernels: build f16 m_norm / m_norm^T / norms in ws ----------------
__global__ void prep_norm_kernel(const float* __restrict__ mem, float* __restrict__ g_norm) {
  int t = threadIdx.x;
  if (t < NPAD) {
    float nrm = 1.0f;
    if (t < N_MEM) {
      const float* r = mem + (size_t)t * C_DIM;
      float ss = 0.0f;
      #pragma unroll 8
      for (int k = 0; k < C_DIM; ++k) { float v = r[k]; ss = fmaf(v, v, ss); }
      nrm = sqrtf(ss);
    }
    g_norm[t] = fmaxf(nrm, 1e-12f);   // NORM_EPS clamp
  }
}

__global__ void prep_fill_kernel(const float* __restrict__ mem, const float* __restrict__ g_norm,
                                 _Float16* __restrict__ g_memn, _Float16* __restrict__ g_memT) {
  const int stride = gridDim.x * blockDim.x;
  const int gid = blockIdx.x * blockDim.x + threadIdx.x;
  // m_norm [NPAD][C_DIM], rows >= N_MEM zero
  for (int i = gid; i < NPAD * C_DIM; i += stride) {
    int n = i / C_DIM, k = i - n * C_DIM;
    float v = 0.0f;
    if (n < N_MEM) v = mem[(size_t)n * C_DIM + k] / g_norm[n];
    g_memn[i] = (_Float16)v;
  }
  // m_norm^T [C_DIM][MT_PITCH], cols >= N_MEM zero (K padding for GEMM2)
  for (int i = gid; i < C_DIM * MT_PITCH; i += stride) {
    int c = i >> 7, n = i & (MT_PITCH - 1);
    float v = 0.0f;
    if (n < N_MEM) v = mem[(size_t)n * C_DIM + c] / g_norm[n];
    g_memT[i] = (_Float16)v;
  }
}

// ---------------- fused main kernel ----------------
__global__ __launch_bounds__(256)
void memmod_main_kernel(const float* __restrict__ F_map,
                        const _Float16* __restrict__ g_memn,
                        const _Float16* __restrict__ g_memT,
                        const float* __restrict__ g_norm,
                        float* __restrict__ F_hat,
                        float* __restrict__ w_out) {
  __shared__ __align__(16) unsigned char s_region[REGION_BYTES];   // scores+zstage, reused as z_hat
  __shared__ __align__(16) _Float16 s_what[M_TILE * WH_PITCH];
  __shared__ float s_norm[NPAD];
  __shared__ float s_red[8 * M_TILE];   // deterministic ||z||^2 reduction

  float*    s_scores = (float*)s_region;
  _Float16* s_zs     = (_Float16*)(s_region + SCORES_BYTES);
  float*    s_zhat   = (float*)s_region;

  const int tid   = threadIdx.x;
  const int lane  = tid & 31;
  const int wv    = tid >> 5;
  const int lm    = lane & 15;
  const int sel8  = (lane >> 4) << 3;   // accumulator row half-select
  const int sel16 = (lane >> 4) << 4;   // B half-select
  const int m0    = wv * 16;

  const int g0 = blockIdx.x * M_TILE;           // first pixel-row of this tile
  const int b  = g0 / P_DIM;
  const int p0 = g0 - b * P_DIM;
  const size_t fbase = (size_t)b * (C_DIM * P_DIM) + p0;

  if (tid < NPAD) s_norm[tid] = g_norm[tid];

  float4 r[4];
  float  sq4[4] = {0.f, 0.f, 0.f, 0.f};
  const int pr0 = (4 * tid) & 127;              // the 4 z-rows this thread always stages

  auto stage_load = [&](int s) {
    const int kc0 = s * ZK;
    #pragma unroll
    for (int it = 0; it < 4; ++it) {
      int g = it * 1024 + tid * 4;
      int c = g >> 7;
      int p = g & 127;
      r[it] = *(const float4*)(F_map + fbase + (size_t)(kc0 + c) * P_DIM + p);
    }
  };
  auto stage_store = [&](int buf) {
    _Float16* zb = s_zs + buf * (M_TILE * ZS_PITCH);
    #pragma unroll
    for (int it = 0; it < 4; ++it) {
      int g = it * 1024 + tid * 4;
      int c = g >> 7;
      int p = g & 127;
      zb[(p + 0) * ZS_PITCH + c] = (_Float16)r[it].x;
      zb[(p + 1) * ZS_PITCH + c] = (_Float16)r[it].y;
      zb[(p + 2) * ZS_PITCH + c] = (_Float16)r[it].z;
      zb[(p + 3) * ZS_PITCH + c] = (_Float16)r[it].w;
      sq4[0] = fmaf(r[it].x, r[it].x, sq4[0]);
      sq4[1] = fmaf(r[it].y, r[it].y, sq4[1]);
      sq4[2] = fmaf(r[it].z, r[it].z, sq4[2]);
      sq4[3] = fmaf(r[it].w, r[it].w, sq4[3]);
    }
  };

  // ---- GEMM1: scores = z_f16 @ m_norm^T, K streamed through double-buffered LDS ----
  v8f acc[NT1] = {};
  stage_load(0);
  stage_store(0);
  __syncthreads();

  for (int s = 0; s < NSTAGE; ++s) {
    const int  cur = s & 1;
    const bool pf  = (s + 1) < NSTAGE;
    if (pf) stage_load(s + 1);
    if (s + 2 < NSTAGE) {
      __builtin_prefetch(F_map + fbase + (size_t)((s + 2) * ZK + ((tid * 4) >> 7)) * P_DIM
                           + ((tid * 4) & 127), 0, 1);
    }
    const _Float16* zb = s_zs + cur * (M_TILE * ZS_PITCH);
    v16h a = frag_a(zb + (m0 + lm) * ZS_PITCH + sel16);
    const int k0 = s * ZK;
    #pragma unroll
    for (int t = 0; t < NT1; ++t) {
      v16h bf = frag_b(g_memn + (size_t)(t * 16 + lm) * MN_PITCH + k0 + sel16);
      acc[t] = __builtin_amdgcn_wmma_f32_16x16x32_f16(false, a, false, bf,
                                                      (short)0, acc[t], false, false);
    }
    if (pf) stage_store((s + 1) & 1);
    __syncthreads();
  }

  // dump scores to LDS (C layout: VGPR j -> M=j+8*(lane>=16), N=lane%16)
  #pragma unroll
  for (int t = 0; t < NT1; ++t) {
    float* sr = s_scores + (m0 + sel8) * SC_PITCH + t * 16 + lm;
    #pragma unroll
    for (int j = 0; j < 8; ++j) sr[j * SC_PITCH] = acc[t][j];
  }
  // deterministic ||z||^2 partials
  #pragma unroll
  for (int j = 0; j < 4; ++j) s_red[(tid >> 5) * M_TILE + pr0 + j] = sq4[j];
  __syncthreads();

  // ---- softmax + hard shrinkage, one thread per pixel-row ----
  if (tid < M_TILE) {
    float ss = 0.f;
    #pragma unroll
    for (int u = 0; u < 8; ++u) ss += s_red[u * M_TILE + tid];
    const float sc = (1.0f / fmaxf(sqrtf(ss), 1e-12f)) * TEMP_INV;  // (z/||z||)/TEMP
    float* srow = s_scores + tid * SC_PITCH;
    float mx = -3.402823e38f;
    for (int n = 0; n < N_MEM; ++n) mx = fmaxf(mx, srow[n]);
    float se = 0.f;
    for (int n = 0; n < N_MEM; ++n) {
      float w = expf((srow[n] - mx) * sc);
      srow[n] = w; se += w;
    }
    const float inv = 1.0f / se;
    float s2 = 0.f;
    for (int n = 0; n < N_MEM; ++n) {
      float w = srow[n] * inv;
      float d = w - LAM_C;
      float wh = (d > 0.f) ? (d * w / (d + EPS_C)) : 0.f;
      srow[n] = wh; s2 += wh;
    }
    const float inv2 = 1.0f / fmaxf(s2, EPS_C);
    _Float16* wrow = s_what + tid * WH_PITCH;
    float* gout = w_out + (size_t)(g0 + tid) * N_MEM;
    for (int n = 0; n < N_MEM; ++n) {
      float wf = srow[n] * inv2;
      gout[n] = wf;                                   // output 2: w_hat_spatial
      wrow[n] = (_Float16)(wf * s_norm[n]);           // fold memory row norm into A of GEMM2
    }
    for (int n = N_MEM; n < MT_PITCH; ++n) wrow[n] = (_Float16)0.f;  // K padding
  }
  __syncthreads();

  // ---- GEMM2: z_hat = what @ m_norm (K=128 padded), c processed in 128-chunks ----
  v16h a2[4];
  #pragma unroll
  for (int kq = 0; kq < 4; ++kq)
    a2[kq] = frag_a(s_what + (m0 + lm) * WH_PITCH + kq * 32 + sel16);

  for (int ch = 0; ch < NCHUNK; ++ch) {
    const int c0 = ch * CCHUNK;
    v8f acc2[8] = {};
    #pragma unroll
    for (int kq = 0; kq < 4; ++kq) {
      const int k0 = kq * 32;
      #pragma unroll
      for (int t = 0; t < 8; ++t) {
        v16h bf = frag_b(g_memT + (size_t)(c0 + t * 16 + lm) * MT_PITCH + k0 + sel16);
        acc2[t] = __builtin_amdgcn_wmma_f32_16x16x32_f16(false, a2[kq], false, bf,
                                                         (short)0, acc2[t], false, false);
      }
    }
    // dump transposed into LDS: zhat[c][m], then coalesced float4 stores to F_hat[b][c][p]
    #pragma unroll
    for (int t = 0; t < 8; ++t) {
      float* zr = s_zhat + (t * 16 + lm) * ZH_PITCH + m0 + sel8;
      #pragma unroll
      for (int j = 0; j < 8; ++j) zr[j] = acc2[t][j];
    }
    __syncthreads();
    for (int g = tid; g < CCHUNK * 32; g += 256) {
      int c = g >> 5;
      int q = (g & 31) << 2;
      *(float4*)(F_hat + fbase + (size_t)(c0 + c) * P_DIM + q) =
          *(const float4*)(s_zhat + c * ZH_PITCH + q);
    }
    __syncthreads();
  }
}

// ---------------- launch ----------------
extern "C" void kernel_launch(void* const* d_in, const int* in_sizes, int n_in,
                              void* d_out, int out_size, void* d_ws, size_t ws_size,
                              hipStream_t stream) {
  (void)in_sizes; (void)n_in; (void)out_size; (void)ws_size;
  const float* F_map  = (const float*)d_in[0];
  const float* memory = (const float*)d_in[1];
  float* F_hat = (float*)d_out;
  float* w_out = F_hat + (size_t)B_DIM * C_DIM * P_DIM;   // outputs concatenated flat

  unsigned char* ws = (unsigned char*)d_ws;                // needs ~369 KB of scratch
  _Float16* g_memn = (_Float16*)(ws + WS_MEMN_OFF);
  _Float16* g_memT = (_Float16*)(ws + WS_MEMT_OFF);
  float*    g_norm = (float*)(ws + WS_NORM_OFF);

  prep_norm_kernel<<<1, 128, 0, stream>>>(memory, g_norm);
  prep_fill_kernel<<<180, 256, 0, stream>>>(memory, g_norm, g_memn, g_memT);
  memmod_main_kernel<<<ROWS_T / M_TILE, 256, 0, stream>>>(F_map, g_memn, g_memT, g_norm,
                                                          F_hat, w_out);
}
